// NodeProcessor_16415365006069
// MI455X (gfx1250) — matrix operand order repaired and checked by reading, not compiled, not run on hardware
//
#include <hip/hip_runtime.h>
#include <hip/hip_bf16.h>

// ---------------- problem constants ----------------
constexpr int N_NODES = 50000;
constexpr int N_EDGES = 1600000;
constexpr int D       = 128;   // node/edge feature dim
constexpr int D_HID   = 256;   // hidden dim
constexpr int K1      = 2 * D; // 256, GEMM1 K
constexpr int KC1     = K1 / 32;    // 8 k-chunks
constexpr int NT1     = D_HID / 16; // 16 n-tiles
constexpr int KC2     = D_HID / 32; // 8
constexpr int NT2     = D / 16;     // 8
constexpr int NTILES  = N_NODES / 16; // 3125 (exact)

typedef __attribute__((ext_vector_type(16))) __bf16 v16bf;
typedef __attribute__((ext_vector_type(8)))  float  v8f;

// ---------------- workspace layout ----------------
// [0, AGG_BYTES)          : agg  float[N_NODES*D]
// [AGG_BYTES, +128KB)     : packed W1 bf16 [NT1][KC1][32][16]
// then                    : packed W2 bf16 [NT2][KC2][32][16]
constexpr size_t AGG_BYTES = (size_t)N_NODES * D * sizeof(float); // 25.6 MB
constexpr size_t PW1_ELEMS = (size_t)NT1 * KC1 * 32 * 16;         // 65536
constexpr size_t PW2_ELEMS = (size_t)NT2 * KC2 * 32 * 16;         // 32768

// ---------------- kernel 1: zero agg ----------------
__global__ void zero_agg_kernel(float4* __restrict__ agg4, int n4) {
    int i = blockIdx.x * blockDim.x + threadIdx.x;
    if (i < n4) agg4[i] = make_float4(0.f, 0.f, 0.f, 0.f);
}

// ---------------- kernels 2/3: pack weights into WMMA B layout ----------------
// B operand (32x16 bf16, KxN): lane holds N = lane&15, K = (e&7) + 8*(lane>>4) + 16*(e>>3)
// within each 32-K chunk; 16 contiguous bf16 per lane.
__global__ void pack_w_kernel(const float* __restrict__ W, __bf16* __restrict__ outp,
                              int ncols, int total) {
    int idx = blockIdx.x * blockDim.x + threadIdx.x;
    if (idx >= total) return;
    int e    = idx & 15;
    int lane = (idx >> 4) & 31;
    int c    = (idx >> 9) & 7;   // k-chunk (8 for both GEMMs)
    int t    = idx >> 12;        // n-tile
    int k = c * 32 + ((e >> 3) * 16) + ((lane >> 4) * 8) + (e & 7);
    int n = t * 16 + (lane & 15);
    outp[idx] = (__bf16)W[(size_t)k * ncols + n];
}

// ---------------- kernel 4: scatter-add edges -> agg ----------------
// one thread per float4 of an edge feature row; hardware f32 atomics (L2-resident agg)
__global__ void scatter_kernel(const float* __restrict__ edge_attr,
                               const int*  __restrict__ edge_index,
                               float* __restrict__ agg) {
    long long i = (long long)blockIdx.x * blockDim.x + threadIdx.x;
    if (i >= (long long)N_EDGES * (D / 4)) return;
    int e = (int)(i >> 5);          // edge id   (D/4 == 32)
    int q = (int)(i & 31);          // float4 slot within row
    const float4 v = ((const float4*)edge_attr)[(size_t)e * (D / 4) + q];
    int dst = edge_index[N_EDGES + e];           // row 1 = destinations
    float* p = agg + (size_t)dst * D + q * 4;
    __hip_atomic_fetch_add(p + 0, v.x, __ATOMIC_RELAXED, __HIP_MEMORY_SCOPE_AGENT);
    __hip_atomic_fetch_add(p + 1, v.y, __ATOMIC_RELAXED, __HIP_MEMORY_SCOPE_AGENT);
    __hip_atomic_fetch_add(p + 2, v.z, __ATOMIC_RELAXED, __HIP_MEMORY_SCOPE_AGENT);
    __hip_atomic_fetch_add(p + 3, v.w, __ATOMIC_RELAXED, __HIP_MEMORY_SCOPE_AGENT);
}

// ---------------- kernel 5: fused MLP + LayerNorm + residual ----------------
// 128 threads = 4 waves; each wave computes one 16-node tile with bf16 WMMA.
__global__ __launch_bounds__(128) void fused_mlp_kernel(
    const float* __restrict__ x, const float* __restrict__ agg,
    const __bf16* __restrict__ pw1, const __bf16* __restrict__ pw2,
    const float* __restrict__ b1, const float* __restrict__ b2,
    const float* __restrict__ gamma, const float* __restrict__ beta,
    float* __restrict__ out)
{
    // per-wave staging buffer: h1 as bf16[16][256] (8KB) reused as h2 float[16][128] (8KB)
    __shared__ __align__(16) char smem[4][16 * 256 * 2];
    __shared__ float s_mu[4][16];
    __shared__ float s_rs[4][16];

    const int lane = threadIdx.x & 31;
    const int wave = threadIdx.x >> 5;
    int tile = blockIdx.x * 4 + wave;
    if (tile >= NTILES) tile = NTILES - 1;  // uniform clamp: redundant identical work

    const int m    = lane & 15;   // A-operand row owned by this lane
    const int khi  = lane >> 4;   // K-half select (A and B layouts)
    const int node0 = tile * 16;
    const size_t rowA = (size_t)(node0 + m) * D;

    // ---- load A (16 x 256) in WMMA A layout, cast fp32 -> bf16 ----
    // lane(m,khi), element e: K = (e&7) + 8*khi + 16*(e>>3) within each 32-chunk
    v16bf A[KC1];
#pragma unroll
    for (int c = 0; c < KC1; ++c) {
#pragma unroll
        for (int e = 0; e < 8; ++e) {
            int k0 = c * 32 + 8 * khi + e;        // e in [0,8)
            int k1 = c * 32 + 16 + 8 * khi + e;   // e in [8,16)
            float f0 = (k0 < D) ? x[rowA + k0] : agg[rowA + (k0 - D)];
            float f1 = (k1 < D) ? x[rowA + k1] : agg[rowA + (k1 - D)];
            A[c][e]     = (__bf16)f0;
            A[c][e + 8] = (__bf16)f1;
        }
    }

    __bf16* h1 = (__bf16*)smem[wave];
    float*  h2 = (float*)smem[wave];

    // ---- GEMM1: H1 = silu(A @ W1 + b1), staged to LDS as bf16 ----
#pragma unroll 1
    for (int t = 0; t < NT1; ++t) {
        v8f acc = {};
        const __bf16* wp = pw1 + (size_t)t * KC1 * 32 * 16;
#pragma unroll
        for (int c = 0; c < KC1; ++c) {
            v16bf B = *(const v16bf*)(wp + ((size_t)c * 32 + lane) * 16);
            acc = __builtin_amdgcn_wmma_f32_16x16x32_bf16(
                false, A[c], false, B, (short)0, acc, false, false);
        }
        // C layout: row = v + 8*khi, col = t*16 + (lane&15)
        float bias = b1[t * 16 + (lane & 15)];
#pragma unroll
        for (int v = 0; v < 8; ++v) {
            float a = acc[v] + bias;
            float s = a / (1.0f + __expf(-a));       // silu
            int row = v + 8 * khi;
            h1[row * D_HID + t * 16 + (lane & 15)] = (__bf16)s;
        }
    }
    __syncthreads();

    // ---- re-load H1 from LDS in A layout for GEMM2 ----
    v16bf A2[KC2];
#pragma unroll
    for (int c = 0; c < KC2; ++c) {
#pragma unroll
        for (int e = 0; e < 8; ++e) {
            A2[c][e]     = h1[m * D_HID + c * 32 + 8 * khi + e];
            A2[c][e + 8] = h1[m * D_HID + c * 32 + 16 + 8 * khi + e];
        }
    }
    __syncthreads();   // everyone has read h1; safe to overwrite with h2

    // ---- GEMM2: H2 = H1 @ W2 + b2, staged to LDS as fp32 ----
#pragma unroll 1
    for (int t = 0; t < NT2; ++t) {
        v8f acc = {};
        const __bf16* wp = pw2 + (size_t)t * KC2 * 32 * 16;
#pragma unroll
        for (int c = 0; c < KC2; ++c) {
            v16bf B = *(const v16bf*)(wp + ((size_t)c * 32 + lane) * 16);
            acc = __builtin_amdgcn_wmma_f32_16x16x32_bf16(
                false, A2[c], false, B, (short)0, acc, false, false);
        }
        float bias = b2[t * 16 + (lane & 15)];
#pragma unroll
        for (int v = 0; v < 8; ++v) {
            int row = v + 8 * khi;
            h2[row * D + t * 16 + (lane & 15)] = acc[v] + bias;
        }
    }
    __syncthreads();

    // ---- LayerNorm stats: lanes 0..15 each own one row of 128 ----
    if (lane < 16) {
        const float* hr = h2 + lane * D;
        float s = 0.f, ss = 0.f;
#pragma unroll
        for (int j = 0; j < D; ++j) { float v = hr[j]; s += v; ss += v * v; }
        float mu  = s * (1.0f / D);
        float var = ss * (1.0f / D) - mu * mu;
        s_mu[wave][lane] = mu;
        s_rs[wave][lane] = rsqrtf(var + 1e-5f);
    }
    __syncthreads();

    // ---- normalize + affine + residual, coalesced writes ----
#pragma unroll 4
    for (int it = 0; it < (16 * D) / 32; ++it) {
        int f   = it * 32 + lane;
        int row = f >> 7;
        int col = f & (D - 1);
        float v = (h2[f] - s_mu[wave][row]) * s_rs[wave][row];
        float o = v * gamma[col] + beta[col] + x[(size_t)(node0 + row) * D + col];
        out[(size_t)(node0 + row) * D + col] = o;
    }
}

// ---------------- host launcher ----------------
extern "C" void kernel_launch(void* const* d_in, const int* in_sizes, int n_in,
                              void* d_out, int out_size, void* d_ws, size_t ws_size,
                              hipStream_t stream) {
    (void)in_sizes; (void)n_in; (void)out_size; (void)ws_size;
    const float* x     = (const float*)d_in[0];
    const int*   ei    = (const int*)d_in[1];   // [2, N_EDGES] int32
    const float* ea    = (const float*)d_in[2];
    const float* W1    = (const float*)d_in[3];
    const float* b1    = (const float*)d_in[4];
    const float* W2    = (const float*)d_in[5];
    const float* b2    = (const float*)d_in[6];
    const float* gamma = (const float*)d_in[7];
    const float* beta  = (const float*)d_in[8];
    float* out = (float*)d_out;

    char* ws = (char*)d_ws;
    float*  agg = (float*)ws;
    __bf16* pw1 = (__bf16*)(ws + AGG_BYTES);
    __bf16* pw2 = (__bf16*)(ws + AGG_BYTES + PW1_ELEMS * sizeof(__bf16));

    // 1) zero agg
    {
        int n4 = N_NODES * D / 4;
        zero_agg_kernel<<<(n4 + 255) / 256, 256, 0, stream>>>((float4*)agg, n4);
    }
    // 2/3) pack weights into WMMA B layout (bf16)
    pack_w_kernel<<<((int)PW1_ELEMS + 255) / 256, 256, 0, stream>>>(W1, pw1, D_HID, (int)PW1_ELEMS);
    pack_w_kernel<<<((int)PW2_ELEMS + 255) / 256, 256, 0, stream>>>(W2, pw2, D,     (int)PW2_ELEMS);
    // 4) scatter-add edge features
    {
        long long total = (long long)N_EDGES * (D / 4);
        int blocks = (int)((total + 255) / 256);
        scatter_kernel<<<blocks, 256, 0, stream>>>(ea, ei, agg);
    }
    // 5) fused MLP + LN + residual (4 tiles of 16 nodes per block)
    {
        int blocks = (NTILES + 3) / 4;
        fused_mlp_kernel<<<blocks, 128, 0, stream>>>(x, agg, pw1, pw2, b1, b2, gamma, beta, out);
    }
}
